// HilbertRingDilatedAttention_51531017617705
// MI455X (gfx1250) — matrix operand order repaired and checked by reading, not compiled, not run on hardware
//
#include <hip/hip_runtime.h>

typedef __attribute__((ext_vector_type(16))) _Float16 v16h;
typedef __attribute__((ext_vector_type(8)))  float    v8f;

#define NSEQ  2048
#define NHEAD 16
#define DHEAD 64
#define HD    (NHEAD * DHEAD)

// Q scale: 1/sqrt(64) * log2(e)  (softmax computed in base-2 domain)
#define QSCALE 0.180336880f

// Boustrophedon "hilbert" map for N=2048 (g=64): involution.
__device__ __forceinline__ int smap(int p) {
  const int c = p & 63;
  return ((p >> 6) & 1) ? (p + 63 - 2 * c) : p;
}

// One 32-key flash-attention tile. MASK=true only for the diagonal tile.
template <bool MASK>
__device__ __forceinline__ void attn_tile(int m0, int n0, int lcol, int hi,
                                          int dg, int dmask,
                                          const float* __restrict__ kb,
                                          const float* __restrict__ vb0,
                                          const v16h (&qb)[2][2],
                                          v8f (&acc)[2][4],
                                          float (&m_run)[2], float (&l_run)[2]) {
  // ---- S^T = K_tile x Q^T : K loaded once, used by both query tiles.
  v8f st[2][2] = {};
  #pragma unroll
  for (int s = 0; s < 2; ++s) {
    const int    mrow = m0 + s * 16 + lcol;
    const int    krow = smap(((mrow & dmask) << dg) + dg);
    const float* kp   = kb + (size_t)krow * HD;
    #pragma unroll
    for (int c = 0; c < 2; ++c) {
      v16h ka;                          // A layout 16x32 f16 (contiguous f4 loads)
      const float* g1 = kp + c * 32 + hi * 16;
      #pragma unroll
      for (int j = 0; j < 2; ++j) {
        float4 f1 = *(const float4*)(g1 + 4 * j);
        ka[4 * j + 0] = (_Float16)f1.x;  ka[4 * j + 1] = (_Float16)f1.y;
        ka[4 * j + 2] = (_Float16)f1.z;  ka[4 * j + 3] = (_Float16)f1.w;
        float4 f2 = *(const float4*)(g1 + 8 + 4 * j);
        ka[8 + 4 * j + 0] = (_Float16)f2.x;  ka[8 + 4 * j + 1] = (_Float16)f2.y;
        ka[8 + 4 * j + 2] = (_Float16)f2.z;  ka[8 + 4 * j + 3] = (_Float16)f2.w;
      }
      #pragma unroll
      for (int t = 0; t < 2; ++t)
        st[t][s] = __builtin_amdgcn_wmma_f32_16x16x32_f16(false, ka, false, qb[t][c],
                                                          (short)0, st[t][s], false, false);
    }
  }

  // ---- Online softmax (base-2) per query tile; lane owns row n = n0+16t+lcol.
  v16h pa[2];
  #pragma unroll
  for (int t = 0; t < 2; ++t) {
    float sv[16];
    #pragma unroll
    for (int s = 0; s < 2; ++s)
      #pragma unroll
      for (int g = 0; g < 8; ++g) {
        float x = st[t][s][g];
        if (MASK) {                     // diagonal tile only: m_abs > n_abs
          if ((m0 + s * 16 + g + 8 * hi) > (n0 + 16 * t + lcol)) x = -1.0e30f;
        }
        sv[s * 8 + g] = x;
      }
    float tmax = sv[0];
    #pragma unroll
    for (int i = 1; i < 16; ++i) tmax = fmaxf(tmax, sv[i]);
    tmax = fmaxf(tmax, __shfl_xor(tmax, 16, 32));
    const float newm  = fmaxf(m_run[t], tmax);
    const float alpha = __builtin_amdgcn_exp2f(m_run[t] - newm);
    float psum = 0.0f;
    float pv[16];
    #pragma unroll
    for (int i = 0; i < 16; ++i) {
      pv[i] = __builtin_amdgcn_exp2f(sv[i] - newm);
      psum += pv[i];
    }
    psum += __shfl_xor(psum, 16, 32);
    l_run[t] = l_run[t] * alpha + psum;
    m_run[t] = newm;

    // Rescale accumulator: O row M = g + 8*hi's stats live at lane g + hi*24.
    #pragma unroll
    for (int g = 0; g < 8; ++g) {
      const float ag = __shfl(alpha, g + hi * 24, 32);
      #pragma unroll
      for (int c = 0; c < 4; ++c) acc[t][c][g] *= ag;
    }

    // P^T scores are already in A-layout of P (16n x 32m): zero shuffles.
    #pragma unroll
    for (int g = 0; g < 8; ++g) {
      pa[t][g]     = (_Float16)pv[g];      // sub-tile 0: m = g + 8*hi
      pa[t][8 + g] = (_Float16)pv[8 + g];  // sub-tile 1: m = 16 + g + 8*hi
    }
  }

  // ---- V as B-operand (32m x 16d per d-chunk), shared by both query tiles.
  int voff[16];
  #pragma unroll
  for (int j = 0; j < 16; ++j) {
    const int m = m0 + (j & 7) + 8 * hi + 16 * (j >> 3);
    voff[j] = smap(((m & dmask) << dg) + dg) * HD;
  }
  #pragma unroll
  for (int c = 0; c < 4; ++c) {
    v16h vbm;
    const float* vp = vb0 + c * 16;
    #pragma unroll
    for (int j = 0; j < 16; ++j) vbm[j] = (_Float16)vp[voff[j]];
    #pragma unroll
    for (int t = 0; t < 2; ++t)
      acc[t][c] = __builtin_amdgcn_wmma_f32_16x16x32_f16(false, pa[t], false, vbm,
                                                         (short)0, acc[t][c], false, false);
  }
}

__global__ __launch_bounds__(256)
void hilbert_attn_wmma(const float* __restrict__ q,
                       const float* __restrict__ k,
                       const float* __restrict__ v,
                       const int*   __restrict__ is_causal,
                       float*       __restrict__ out) {
  const int lane = threadIdx.x & 31;
  const int wib  = threadIdx.x >> 5;
  const int gw   = blockIdx.x * (blockDim.x >> 5) + wib; // global wave id
  const int qt   = gw & 63;             // 64 query blocks of 32 rows
  const int h    = (gw >> 6) & 15;      // head
  const int b    = gw >> 10;            // batch
  const int n0   = qt << 5;             // first query row of this wave

  const int lcol = lane & 15;           // N-column of B/C/D layouts
  const int hi   = lane >> 4;           // half-wave selector

  // Branchless, wave-uniform dilation parameters:
  // heads 0-4: m ; 5-9: 2*(m&1023)+1 ; 10-15: 4*(m&511)+2
  const int dg    = (h >= 5) + (h >= 10);
  const int dmask = 2047 >> dg;

  const int causal = *is_causal;
  const size_t bbase = (size_t)b * NSEQ * HD + (size_t)h * DHEAD;
  const float* kb  = k + bbase;
  const float* vb0 = v + bbase + lcol;

  // ---- Load both 16-row Q tiles once as B-operands of S^T = K x Q^T.
  // B layout 32x16 f16: lane holds kk = c*32 + 16*hi + (0..15), contiguous.
  v16h qb[2][2];
  #pragma unroll
  for (int t = 0; t < 2; ++t) {
    const float* qp = q + bbase + (size_t)smap(n0 + 16 * t + lcol) * HD;
    #pragma unroll
    for (int c = 0; c < 2; ++c) {
      const float* p0 = qp + c * 32 + hi * 16;
      #pragma unroll
      for (int j = 0; j < 4; ++j) {
        float4 f = *(const float4*)(p0 + 4 * j);
        qb[t][c][4 * j + 0] = (_Float16)(f.x * QSCALE);
        qb[t][c][4 * j + 1] = (_Float16)(f.y * QSCALE);
        qb[t][c][4 * j + 2] = (_Float16)(f.z * QSCALE);
        qb[t][c][4 * j + 3] = (_Float16)(f.w * QSCALE);
      }
    }
  }

  v8f  acc[2][4] = {};                  // two 16x64 O tiles, C/D layout
  float m_run[2] = {-INFINITY, -INFINITY};
  float l_run[2] = {0.0f, 0.0f};

  // ---- Steady-state loop: every tile strictly below the diagonal, no masking.
  const int mMain = causal ? n0 : NSEQ;
  for (int m0 = 0; m0 < mMain; m0 += 32) {
    #pragma unroll
    for (int s = 0; s < 2; ++s) {       // prefetch next tile's K rows (L2 warm)
      const int pn = m0 + 32 + s * 16 + lcol;
      __builtin_prefetch(kb + (size_t)smap(((pn & dmask) << dg) + dg) * HD + hi * 32, 0, 1);
    }
    attn_tile<false>(m0, n0, lcol, hi, dg, dmask, kb, vb0, qb, acc, m_run, l_run);
  }
  // ---- Peeled diagonal tile: the only one that needs the causal mask.
  if (causal)
    attn_tile<true>(n0, n0, lcol, hi, dg, dmask, kb, vb0, qb, acc, m_run, l_run);

  // ---- Epilogue: normalize rows, scatter back through S (involution).
  #pragma unroll
  for (int t = 0; t < 2; ++t)
    #pragma unroll
    for (int g = 0; g < 8; ++g) {
      const float lr  = __shfl(l_run[t], g + hi * 24, 32);
      const float inv = 1.0f / lr;
      const int  orow = smap(n0 + 16 * t + g + 8 * hi);
      float* op = out + bbase + (size_t)orow * HD + lcol;
      #pragma unroll
      for (int c = 0; c < 4; ++c) op[c * 16] = acc[t][c][g] * inv;
    }
}

extern "C" void kernel_launch(void* const* d_in, const int* in_sizes, int n_in,
                              void* d_out, int out_size, void* d_ws, size_t ws_size,
                              hipStream_t stream) {
  const float* q   = (const float*)d_in[0];
  const float* k   = (const float*)d_in[1];
  const float* v   = (const float*)d_in[2];
  const int*   isc = (const int*)d_in[3];
  float* out = (float*)d_out;

  const int waves = 2 * NHEAD * (NSEQ / 32);   // 2048 waves, 1 per (b,h,32-row block)
  dim3 block(256);                              // 8 waves/block, no LDS
  dim3 grid(waves / 8);
  hipLaunchKernelGGL(hilbert_attn_wmma, grid, block, 0, stream, q, k, v, isc, out);
}
